// RTNGemmLinear_72997264163016
// MI455X (gfx1250) — matrix operand order repaired and checked by reading, not compiled, not run on hardware
//
#include <hip/hip_runtime.h>

typedef __attribute__((ext_vector_type(16))) _Float16 v16h;
typedef __attribute__((ext_vector_type(8)))  _Float16 v8h;
typedef __attribute__((ext_vector_type(8)))  float    v8f;
typedef __attribute__((ext_vector_type(4)))  int      v4i;

#define IN_F    4096
#define OUT_F   11008
#define NTOK    16
#define GROUP   128
#define NGROUPS (IN_F / GROUP)     // 32
#define NTILES  (OUT_F / 16)       // 688
#define WPB     8                  // waves per block (wave32)
#define KCHUNK  (IN_F / WPB)       // 512 per wave
#define KITERS  (KCHUNK / 32)      // 16 WMMA steps per wave

// ---- prep: convert x (fp32) -> f16 once into workspace (L2-resident reuse) ----
__global__ void rtn_xcvt(const float* __restrict__ x, _Float16* __restrict__ xh) {
  int i = blockIdx.x * blockDim.x + threadIdx.x;
  if (i < NTOK * IN_F) xh[i] = (_Float16)x[i];
}

// ---- main: one 16x16 output tile per block, K split 8-ways across the block ----
__global__ __launch_bounds__(256, 2)
void rtn_wmma_gemm(const _Float16* __restrict__ xh,
                   const int* __restrict__ qw,
                   const float* __restrict__ sc,
                   const float* __restrict__ bias,
                   float* __restrict__ out) {
  __shared__ float red[WPB][256];

  const int tid  = threadIdx.x;
  const int wave = tid >> 5;
  const int lane = tid & 31;
  const int n    = lane & 15;   // B column within tile / A row (token)
  const int h    = lane >> 4;   // half-wave selector (K sub-block)
  const int nbase  = blockIdx.x << 4;
  const int row    = nbase + n;              // weight row (output feature)
  const int kstart = wave * KCHUNK;

  const _Float16* xrow = xh + n * IN_F;
  const int*      qrow = qw + (size_t)row * IN_F;
  const float*    srow = sc + row * NGROUPS;

  v8f c = {};

  // double-buffered pipeline registers
  v4i   qbuf[2][4];
  v8h   abuf[2][2];
  float sbuf[2];

  auto stage = [&](int it, int buf) {
    const int k0 = kstart + (it << 5);
    const int kb = k0 + (h << 4);                  // this lane's 16-code chunk
    const v4i* qp = (const v4i*)(qrow + kb);
    qbuf[buf][0] = __builtin_nontemporal_load(qp + 0);   // stream qweight: no reuse
    qbuf[buf][1] = __builtin_nontemporal_load(qp + 1);
    qbuf[buf][2] = __builtin_nontemporal_load(qp + 2);
    qbuf[buf][3] = __builtin_nontemporal_load(qp + 3);
    abuf[buf][0] = *(const v8h*)(xrow + k0 + (h << 4));        // A: K = 16h+0..7
    abuf[buf][1] = *(const v8h*)(xrow + k0 + 8 + (h << 4));    // A: K = 16h+8..15
    sbuf[buf] = srow[kb >> 7];                     // one scale per 16-code chunk
  };

  auto step = [&](int buf) {
    const float s = sbuf[buf];
    const float z = -8.0f * s;                     // symmetric RTN zero-point
    v16h b;
#pragma unroll
    for (int e = 0; e < 4; ++e) {
      v4i q = qbuf[buf][e];
#pragma unroll
      for (int j = 0; j < 4; ++j)
        b[e * 4 + j] = (_Float16)__builtin_fmaf((float)q[j], s, z);
    }
    union { v16h v; v8h half8[2]; } a;
    a.half8[0] = abuf[buf][0];
    a.half8[1] = abuf[buf][1];
    c = __builtin_amdgcn_wmma_f32_16x16x32_f16(false, a.v, false, b,
                                               (short)0, c, false, false);
  };

  stage(0, 0);
#pragma unroll
  for (int it = 0; it < KITERS; ++it) {
    if (it + 1 < KITERS) stage(it + 1, (it + 1) & 1);  // prefetch next K-step
    step(it & 1);
  }

  // deterministic cross-wave K reduction through LDS (kept in C-layout space)
#pragma unroll
  for (int r = 0; r < 8; ++r) red[wave][r * 32 + lane] = c[r];
  __syncthreads();

  float sum = 0.0f;
#pragma unroll
  for (int w = 0; w < WPB; ++w) sum += red[w][tid];

  // map C layout -> (token M, feature N): VGPR r, lane l: M = r + 8*(l>=16), N = l%16
  const int r = tid >> 5;
  const int l = tid & 31;
  const int M = r + ((l >> 4) << 3);
  const int N = l & 15;
  out[(size_t)M * OUT_F + nbase + N] = sum + bias[nbase + N];
}

extern "C" void kernel_launch(void* const* d_in, const int* in_sizes, int n_in,
                              void* d_out, int out_size, void* d_ws, size_t ws_size,
                              hipStream_t stream) {
  const float* x    = (const float*)d_in[0];   // [16, 4096] fp32
  const int*   qw   = (const int*)d_in[1];     // [11008, 4096] int32 codes 0..15
  const float* sc   = (const float*)d_in[2];   // [11008, 32] fp32
  const float* bias = (const float*)d_in[3];   // [11008] fp32
  float* out  = (float*)d_out;                 // [16, 11008] fp32
  _Float16* xh = (_Float16*)d_ws;              // 128 KB scratch: x in f16

  rtn_xcvt<<<(NTOK * IN_F + 255) / 256, 256, 0, stream>>>(x, xh);
  rtn_wmma_gemm<<<NTILES, 256, 0, stream>>>(xh, qw, sc, bias, out);
}
